// MultiQueryAttention_44418551775724
// MI455X (gfx1250) — matrix operand order repaired and checked by reading, not compiled, not run on hardware
//
#include <hip/hip_runtime.h>
#include <hip/hip_bf16.h>

typedef __bf16 bf16_t;
typedef __attribute__((ext_vector_type(16))) __bf16 bf16x16;
typedef __attribute__((ext_vector_type(8)))  __bf16 bf16x8;
typedef __attribute__((ext_vector_type(4)))  __bf16 bf16x4;
typedef __attribute__((ext_vector_type(2)))  __bf16 bf16x2;
typedef __attribute__((ext_vector_type(8)))  float  f32x8;
typedef __attribute__((ext_vector_type(4)))  float  f32x4;

// ---------- bf16 conversion ----------
#if __has_builtin(__builtin_amdgcn_cvt_pk_bf16_f32)
__device__ __forceinline__ bf16x2 f2bf_pk(float a, float b) {
    return __builtin_amdgcn_cvt_pk_bf16_f32(a, b);
}
__device__ __forceinline__ bf16_t f2bf(float f) { return f2bf_pk(f, f)[0]; }
#else
// plain cast: clang lowers to native cvt if the target has it
__device__ __forceinline__ bf16_t f2bf(float f) { return (bf16_t)f; }
__device__ __forceinline__ bf16x2 f2bf_pk(float a, float b) {
    bf16x2 r; r[0] = (bf16_t)a; r[1] = (bf16_t)b; return r;
}
#endif

__device__ __forceinline__ bf16x4 f2bf4(f32x4 v) {
    bf16x2 lo = f2bf_pk(v[0], v[1]);
    bf16x2 hi = f2bf_pk(v[2], v[3]);
    bf16x4 r; r[0] = lo[0]; r[1] = lo[1]; r[2] = hi[0]; r[3] = hi[1];
    return r;
}

__device__ __forceinline__ f32x8 wmma_bf16(bf16x16 a, bf16x16 b, f32x8 c) {
    return __builtin_amdgcn_wmma_f32_16x16x32_bf16(
        false, a, false, b, (short)0, c, false, false);
}

// A-fragment (16x32 bf16, MxK): lane<16: row=lane, K={0..7,16..23};
// lane>=16: row=lane-16, K={8..15,24..31}. base -> [row0][k0], row-major.
__device__ __forceinline__ bf16x16 load_a_frag(const bf16_t* base, int stride) {
    int lane = threadIdx.x & 31;
    const bf16_t* p = base + (lane & 15) * stride + ((lane >> 4) << 3);
    bf16x8 lo = *reinterpret_cast<const bf16x8*>(p);
    bf16x8 hi = *reinterpret_cast<const bf16x8*>(p + 16);
    bf16x16 r;
#pragma unroll
    for (int e = 0; e < 8; ++e) { r[e] = lo[e]; r[e + 8] = hi[e]; }
    return r;
}

// B-fragment (32x16 bf16, KxN): lane<16: col=lane, K={0..7,16..23}; lane>=16:
// col=lane-16, K={8..15,24..31}. base -> N-major layout [n][k] (col's K contiguous).
__device__ __forceinline__ bf16x16 load_b_frag(const bf16_t* base, int stride) {
    int lane = threadIdx.x & 31;
    const bf16_t* p = base + (lane & 15) * stride + ((lane >> 4) << 3);
    bf16x8 lo = *reinterpret_cast<const bf16x8*>(p);
    bf16x8 hi = *reinterpret_cast<const bf16x8*>(p + 16);
    bf16x16 r;
#pragma unroll
    for (int e = 0; e < 8; ++e) { r[e] = lo[e]; r[e + 8] = hi[e]; }
    return r;
}

// ---------------------------------------------------------------
// GEMM: C[M,N] = A[M,K] @ B[K,N], bf16 WMMA, fp32 out.
// Block tile 128x128, BK=32, 8 waves in 2(M)x4(N), LDS double-buffered.
// AF32: A operand is fp32 (convert) or bf16 (copy).
// ---------------------------------------------------------------
template <int N, int K, bool AF32>
__global__ __launch_bounds__(256)
void k_gemm(const void* __restrict__ Av, const float* __restrict__ B,
            float* __restrict__ C) {
    __shared__ __align__(16) bf16_t Alds[2][128 * 40];
    __shared__ __align__(16) bf16_t Blds[2][128 * 40];

    const int tid = threadIdx.x, wave = tid >> 5, lane = tid & 31;
    const int wm = wave >> 2, wn = wave & 3;
    const int rowBase = blockIdx.y * 128, colBase = blockIdx.x * 128;
    const int half = lane >> 4, cn = lane & 15;

    // per-thread staging coordinates
    const int am0 = tid >> 3, aq4 = (tid & 7) << 2;   // A: rows of 32 elems
    const int bk0 = tid >> 5, bn4 = (tid & 31) << 2;  // B: 8 k-rows apart per it

    const float*  Af = (const float*)Av;
    const bf16_t* Ab = (const bf16_t*)Av;

    f32x8 acc[4][2];
    const f32x8 z = {0.f,0.f,0.f,0.f,0.f,0.f,0.f,0.f};
#pragma unroll
    for (int mi = 0; mi < 4; ++mi)
#pragma unroll
        for (int ni = 0; ni < 2; ++ni) acc[mi][ni] = z;

    f32x4 arf[4]; bf16x8 arb[2]; f32x4 br[4];

    // ---- global loads for one k-tile ----
    auto load_tile = [&](int kk) {
        if (AF32) {
#pragma unroll
            for (int it = 0; it < 4; ++it)
                arf[it] = *reinterpret_cast<const f32x4*>(
                    Af + (size_t)(rowBase + am0 + it * 32) * K + kk + aq4);
        } else {
            const int m0 = tid >> 2, c8 = (tid & 3) << 3;
#pragma unroll
            for (int it = 0; it < 2; ++it)
                arb[it] = *reinterpret_cast<const bf16x8*>(
                    Ab + (size_t)(rowBase + m0 + it * 64) * K + kk + c8);
        }
#pragma unroll
        for (int it = 0; it < 4; ++it)
            br[it] = *reinterpret_cast<const f32x4*>(
                B + (size_t)(kk + bk0 + it * 8) * N + colBase + bn4);
    };
    // ---- store staged tile into LDS buffer ----
    auto store_tile = [&](int buf) {
        if (AF32) {
#pragma unroll
            for (int it = 0; it < 4; ++it)
                *reinterpret_cast<bf16x4*>(&Alds[buf][(am0 + it * 32) * 40 + aq4]) =
                    f2bf4(arf[it]);
        } else {
            const int m0 = tid >> 2, c8 = (tid & 3) << 3;
#pragma unroll
            for (int it = 0; it < 2; ++it)
                *reinterpret_cast<bf16x8*>(&Alds[buf][(m0 + it * 64) * 40 + c8]) =
                    arb[it];
        }
#pragma unroll
        for (int it = 0; it < 4; ++it) {
            bf16x4 w = f2bf4(br[it]);
#pragma unroll
            for (int j = 0; j < 4; ++j)
                Blds[buf][(bn4 + j) * 40 + bk0 + it * 8] = w[j];
        }
    };

    constexpr int T = K / 32;
    load_tile(0);
    store_tile(0);
    __syncthreads();

    for (int t = 0; t < T; ++t) {
        const int cur = t & 1;
        if (t + 1 < T) load_tile((t + 1) * 32);   // prefetch next tile

        bf16x16 af[4], bfr[2];
#pragma unroll
        for (int mi = 0; mi < 4; ++mi)
            af[mi] = load_a_frag(&Alds[cur][(wm * 64 + mi * 16) * 40], 40);
#pragma unroll
        for (int ni = 0; ni < 2; ++ni)
            bfr[ni] = load_b_frag(&Blds[cur][(wn * 32 + ni * 16) * 40], 40);
#pragma unroll
        for (int mi = 0; mi < 4; ++mi)
#pragma unroll
            for (int ni = 0; ni < 2; ++ni)
                acc[mi][ni] = wmma_bf16(af[mi], bfr[ni], acc[mi][ni]);

        if (t + 1 < T) store_tile(1 - cur);
        __syncthreads();
    }

    // epilogue: single base, compile-time store offsets
    float* p = C + (size_t)(rowBase + wm * 64 + half * 8) * N
                 + colBase + wn * 32 + cn;
#pragma unroll
    for (int mi = 0; mi < 4; ++mi)
#pragma unroll
        for (int ni = 0; ni < 2; ++ni)
#pragma unroll
            for (int i = 0; i < 8; ++i)
                p[(size_t)(mi * 16 + i) * N + ni * 16] = acc[mi][ni][i];
}

// ---------------------------------------------------------------
// RoPE + split/reorg. xqkv[4096][3072] ->
//   Q bf16 [B][16][S][128] (rotated), K bf16 [B][4][S][128] (rotated),
//   Vt bf16 [B][4][128][S] (transposed). One block per (b,s) token.
// ---------------------------------------------------------------
__global__ __launch_bounds__(256)
void k_rope(const float* __restrict__ xqkv, bf16_t* __restrict__ Qg,
            bf16_t* __restrict__ Kg, bf16_t* __restrict__ Vt) {
    const int row = blockIdx.x;
    const int b = row >> 11, s = row & 2047;
    const float* xr = xqkv + (size_t)row * 3072;
    const float LOG_THETA = 9.210340371976184f;    // ln(10000)

    for (int p = threadIdx.x; p < 1024; p += 256) {        // Q: 16 heads x 64 pairs
        int h = p >> 6, d = p & 63;
        float fr = __expf(-(float)d * (1.0f / 64.0f) * LOG_THETA);
        float sn, cs; __sincosf((float)s * fr, &sn, &cs);
        float x1 = xr[h * 128 + d], x2 = xr[h * 128 + 64 + d];
        size_t base = ((size_t)(b * 16 + h) * 2048 + s) * 128;
        Qg[base + d]      = f2bf(x1 * cs - x2 * sn);
        Qg[base + 64 + d] = f2bf(x1 * sn + x2 * cs);
    }
    for (int p = threadIdx.x; p < 256; p += 256) {         // K: 4 heads x 64 pairs
        int h = p >> 6, d = p & 63;
        float fr = __expf(-(float)d * (1.0f / 64.0f) * LOG_THETA);
        float sn, cs; __sincosf((float)s * fr, &sn, &cs);
        float x1 = xr[2048 + h * 128 + d], x2 = xr[2048 + h * 128 + 64 + d];
        size_t base = ((size_t)(b * 4 + h) * 2048 + s) * 128;
        Kg[base + d]      = f2bf(x1 * cs - x2 * sn);
        Kg[base + 64 + d] = f2bf(x1 * sn + x2 * cs);
    }
    for (int p = threadIdx.x; p < 512; p += 256) {         // V: 4 heads x 128 dims
        int h = p >> 7, d = p & 127;
        Vt[((size_t)(b * 4 + h) * 128 + d) * 2048 + s] = f2bf(xr[2560 + h * 128 + d]);
    }
}

// ---------------------------------------------------------------
// Causal flash attention.
// Block = (qblk, h, b). 8 waves, wave w owns q rows qbase+16w..+15.
// Key blocks of 64 streamed through LDS. Output: attn bf16 [b*S+s][h*128+d].
// ---------------------------------------------------------------
__global__ __launch_bounds__(256)
void k_attn(const bf16_t* __restrict__ Qg, const bf16_t* __restrict__ Kg,
            const bf16_t* __restrict__ Vt, bf16_t* __restrict__ Og) {
    __shared__ __align__(16) bf16_t Klds[64 * 136];    // [key][d]
    __shared__ __align__(16) bf16_t Vlds[128 * 72];    // [d][key]
    __shared__ __align__(16) bf16_t Plds[8 * 16 * 72]; // per-wave [q][key]

    const int tid = threadIdx.x, wave = tid >> 5, lane = tid & 31;
    const int half = lane >> 4, cn = lane & 15;
    const int b = blockIdx.z, h = blockIdx.y, qblk = blockIdx.x, hkv = h >> 2;
    const int qbase = qblk * 128, qwave = qbase + wave * 16;
    const float scale = 0.08838834764831845f;          // 1/sqrt(128)

    const bf16_t* Qrow = Qg + ((size_t)(b * 16 + h) * 2048 + qwave) * 128;
    bf16x16 qf[4];
#pragma unroll
    for (int c = 0; c < 4; ++c) qf[c] = load_a_frag(Qrow + c * 32, 128);

    float m_i[8], l_i[8];
    f32x8 oacc[8];
    const f32x8 z = {0.f,0.f,0.f,0.f,0.f,0.f,0.f,0.f};
#pragma unroll
    for (int i = 0; i < 8; ++i) { m_i[i] = -1e30f; l_i[i] = 0.f; }
#pragma unroll
    for (int dt = 0; dt < 8; ++dt) oacc[dt] = z;

    const int nkb = 2 * qblk + 2;                      // keys 0 .. qbase+127
    for (int kb = 0; kb < nkb; ++kb) {
        const int kbb = kb * 64;
        const bf16_t* Kgb = Kg + ((size_t)(b * 4 + hkv) * 2048 + kbb) * 128;
        const bf16_t* Vgb = Vt + (size_t)(b * 4 + hkv) * 128 * 2048 + kbb;
#pragma unroll
        for (int it = 0; it < 4; ++it) {               // K block: 64x128
            int idx = tid + it * 256;
            int r = idx >> 4, c8 = (idx & 15) << 3;
            *reinterpret_cast<bf16x8*>(&Klds[r * 136 + c8]) =
                *reinterpret_cast<const bf16x8*>(Kgb + r * 128 + c8);
        }
#pragma unroll
        for (int it = 0; it < 4; ++it) {               // V block: 128x64 (d-major)
            int idx = tid + it * 256;
            int r = idx >> 3, c8 = (idx & 7) << 3;
            *reinterpret_cast<bf16x8*>(&Vlds[r * 72 + c8]) =
                *reinterpret_cast<const bf16x8*>(Vgb + (size_t)r * 2048 + c8);
        }
        __syncthreads();

        const bool skip = (kbb > qwave + 15);          // wave-uniform
        if (!skip) {
            f32x8 sc[4];
#pragma unroll
            for (int kt = 0; kt < 4; ++kt) {
                f32x8 s = z;
#pragma unroll
                for (int c = 0; c < 4; ++c) {
                    bf16x16 bk = load_b_frag(&Klds[(kt * 16) * 136 + c * 32], 136);
                    s = wmma_bf16(qf[c], bk, s);
                }
                sc[kt] = s;
            }
            const bool needMask = (kbb + 63 > qwave);
            float rm[8];
#pragma unroll
            for (int i = 0; i < 8; ++i) rm[i] = -1e30f;
#pragma unroll
            for (int kt = 0; kt < 4; ++kt)
#pragma unroll
                for (int i = 0; i < 8; ++i) {
                    float v = sc[kt][i] * scale;
                    if (needMask) {
                        int kg = kbb + kt * 16 + cn;
                        int qg = qwave + i + half * 8;
                        if (kg > qg) v = -1e30f;
                    }
                    sc[kt][i] = v;
                    rm[i] = fmaxf(rm[i], v);
                }
#pragma unroll
            for (int i = 0; i < 8; ++i) {
                rm[i] = fmaxf(rm[i], __shfl_xor(rm[i], 1, 32));
                rm[i] = fmaxf(rm[i], __shfl_xor(rm[i], 2, 32));
                rm[i] = fmaxf(rm[i], __shfl_xor(rm[i], 4, 32));
                rm[i] = fmaxf(rm[i], __shfl_xor(rm[i], 8, 32));
            }
            float mnew[8], alpha[8], rs[8];
#pragma unroll
            for (int i = 0; i < 8; ++i) {
                mnew[i] = fmaxf(m_i[i], rm[i]);
                alpha[i] = __expf(m_i[i] - mnew[i]);
                m_i[i] = mnew[i];
                rs[i] = 0.f;
            }
#pragma unroll
            for (int kt = 0; kt < 4; ++kt)
#pragma unroll
                for (int i = 0; i < 8; ++i) {
                    float p = __expf(sc[kt][i] - mnew[i]);
                    sc[kt][i] = p;
                    rs[i] += p;
                }
#pragma unroll
            for (int i = 0; i < 8; ++i) {
                rs[i] += __shfl_xor(rs[i], 1, 32);
                rs[i] += __shfl_xor(rs[i], 2, 32);
                rs[i] += __shfl_xor(rs[i], 4, 32);
                rs[i] += __shfl_xor(rs[i], 8, 32);
                l_i[i] = l_i[i] * alpha[i] + rs[i];
            }
#pragma unroll
            for (int dt = 0; dt < 8; ++dt)
#pragma unroll
                for (int i = 0; i < 8; ++i) oacc[dt][i] *= alpha[i];

            bf16_t* Pw = &Plds[wave * 16 * 72];
#pragma unroll
            for (int kt = 0; kt < 4; ++kt)
#pragma unroll
                for (int i = 0; i < 8; ++i)
                    Pw[(i + half * 8) * 72 + kt * 16 + cn] = f2bf(sc[kt][i]);

#pragma unroll
            for (int c = 0; c < 2; ++c) {
                bf16x16 pf = load_a_frag(Pw + c * 32, 72);
#pragma unroll
                for (int dt = 0; dt < 8; ++dt) {
                    bf16x16 bv = load_b_frag(&Vlds[(dt * 16) * 72 + c * 32], 72);
                    oacc[dt] = wmma_bf16(pf, bv, oacc[dt]);
                }
            }
        }
        __syncthreads();
    }
    // epilogue: single base, compile-time offsets
    bf16_t* p = Og + ((size_t)(b * 2048 + qwave + half * 8)) * 2048 + h * 128 + cn;
#pragma unroll
    for (int dt = 0; dt < 8; ++dt)
#pragma unroll
        for (int i = 0; i < 8; ++i)
            p[(size_t)i * 2048 + dt * 16] = f2bf(oacc[dt][i] / l_i[i]);
}

// ---------------------------------------------------------------
extern "C" void kernel_launch(void* const* d_in, const int* in_sizes, int n_in,
                              void* d_out, int out_size, void* d_ws, size_t ws_size,
                              hipStream_t stream) {
    (void)in_sizes; (void)n_in; (void)out_size; (void)ws_size;
    const float* x    = (const float*)d_in[0];   // [2,2048,2048]
    const float* Wqkv = (const float*)d_in[1];   // [2048,3072]
    const float* Wo   = (const float*)d_in[2];   // [2048,2048]
    float* out = (float*)d_out;                  // [2,2048,2048]

    char* ws = (char*)d_ws;
    float*  xqkv = (float*)(ws);                              // 50331648 B
    bf16_t* Qg   = (bf16_t*)(ws + 50331648);                  // 16777216 B
    bf16_t* Kg   = (bf16_t*)(ws + 50331648 + 16777216);       // 4194304 B
    bf16_t* Vt   = (bf16_t*)(ws + 50331648 + 16777216 + 4194304);          // 4194304 B
    bf16_t* attn = (bf16_t*)(ws + 50331648 + 16777216 + 4194304 + 4194304);// 16777216 B

    dim3 g1(3072 / 128, 4096 / 128);
    k_gemm<3072, 2048, true><<<g1, 256, 0, stream>>>(x, Wqkv, xqkv);

    k_rope<<<4096, 256, 0, stream>>>(xqkv, Qg, Kg, Vt);

    dim3 g2(16, 16, 2);
    k_attn<<<g2, 256, 0, stream>>>(Qg, Kg, Vt, attn);

    dim3 g3(2048 / 128, 4096 / 128);
    k_gemm<2048, 2048, false><<<g3, 256, 0, stream>>>(attn, Wo, out);
}
